// TransformerModel_19344532702012
// MI455X (gfx1250) — matrix operand order repaired and checked by reading, not compiled, not run on hardware
//
#include <hip/hip_runtime.h>
#include <math.h>

typedef __attribute__((ext_vector_type(16))) _Float16 v16h;
typedef __attribute__((ext_vector_type(8)))  _Float16 v8h;
typedef __attribute__((ext_vector_type(8)))  float    v8f;

#define B_   16
#define S_   2000
#define IN_  128
#define D_   256
#define H_   2
#define HD_  128
#define FF_  256
#define OUT_ 64
#define M_   (B_ * S_)   // 32000 rows; S_%16==0 so 16-row tiles never cross batches

union U16H { v16h v; v8h h[2]; _Float16 e[16]; };
union U8F  { v8f  v; float    e[8];  };

// ---------------------------------------------------------------------------
// Weight prep: [Nout x K] f32 -> WMMA-fragment-packed f16.
// Packed layout: for k-chunk kc, n-tile tn, element index = ((kc*Ntiles+tn)*32 +
// lane)*16 + i holds W^T[k][n] with k = kc*32 + (lane<16?0:16) + i,
// n = tn*16 + (lane&15)  — contiguous k order, matching the GEMM's A-fragment loads.
// ---------------------------------------------------------------------------
__global__ void wpack_kernel(const float* __restrict__ in, _Float16* __restrict__ out,
                             int Nout, int K, int scale_cols, float scale) {
  int idx = blockIdx.x * blockDim.x + threadIdx.x;
  if (idx >= Nout * K) return;
  int i    = idx & 15;
  int lane = (idx >> 4) & 31;
  int tile = idx >> 9;
  int Ntiles = Nout >> 4;
  int tn = tile % Ntiles;
  int kc = tile / Ntiles;
  int k = kc * 32 + ((lane & 16) ? 16 : 0) + i;
  int n = tn * 16 + (lane & 15);
  float v = in[(size_t)n * K + k];
  if (n < scale_cols) v *= scale;
  out[idx] = (_Float16)v;
}

__global__ void bscale_kernel(const float* __restrict__ in, float* __restrict__ out,
                              int n, int scale_cols, float scale) {
  int i = blockIdx.x * blockDim.x + threadIdx.x;
  if (i < n) out[i] = in[i] * (i < scale_cols ? scale : 1.f);
}

__global__ void f32_to_f16_kernel(const float* __restrict__ in, _Float16* __restrict__ out, int n) {
  int i = blockIdx.x * blockDim.x + threadIdx.x;
  if (i < n) out[i] = (_Float16)in[i];
}

// V transpose for attention: qkv[b*S+s, 512 + h*128 + e] -> vT[((b*H+h)*HD + e)*S + s]
// (write-coalesced; makes P.V B-fragments contiguous 32-byte loads)
__global__ void vtrans_kernel(const _Float16* __restrict__ qkv, _Float16* __restrict__ vT) {
  int idx = blockIdx.x * blockDim.x + threadIdx.x;
  if (idx >= B_ * H_ * HD_ * S_) return;
  int s    = idx % S_;
  int rest = idx / S_;
  int e    = rest % HD_;
  int bh   = rest / HD_;
  int h = bh % H_, b = bh / H_;
  vT[idx] = qkv[((size_t)(b * S_ + s)) * 768 + 512 + h * HD_ + e];
}

// ---------------------------------------------------------------------------
// WMMA GEMM: one wave per 16x64 strip (4 n-tiles share A-fragments).
// C = A[f16, MxK] @ Bpack[f16, fragment-packed KxN]; epilogue: bias,
// optional f32 residual, optional sinusoidal pos-enc, optional relu.
// ---------------------------------------------------------------------------
__global__ __launch_bounds__(32)
void wmma_gemm_kernel(const _Float16* __restrict__ A, const _Float16* __restrict__ Bp,
                      const float* __restrict__ bias, const float* __restrict__ resid,
                      float* __restrict__ out32, _Float16* __restrict__ out16,
                      int K, int N, int relu, int posenc) {
  const int lane = threadIdx.x;
  const int tn0 = blockIdx.x * 4, tm = blockIdx.y;
  const int nh  = lane & 15;
  const int kbA = (lane < 16) ? 0 : 16;       // per-lane K base (contiguous k = 16h + s, same as Bpack)
  const int Ntiles = N >> 4;
  const size_t arow = (size_t)(tm * 16 + nh) * K;

  v8f acc0 = {}, acc1 = {}, acc2 = {}, acc3 = {};
  for (int kk = 0; kk < K; kk += 32) {
    U16H a;
    const _Float16* ap = A + arow + kk + kbA;
    a.h[0] = *(const v8h*)ap;                 // k = kb + 0..7
    a.h[1] = *(const v8h*)(ap + 8);           // k = kb + 8..15
    if (kk + 32 < K) __builtin_prefetch(ap + 32, 0, 0);
    const _Float16* bp = Bp + (((size_t)(kk >> 5) * Ntiles + tn0) << 9) + (lane << 4);
    v16h b0 = *(const v16h*)(bp);
    v16h b1 = *(const v16h*)(bp + 512);
    v16h b2 = *(const v16h*)(bp + 1024);
    v16h b3 = *(const v16h*)(bp + 1536);
    acc0 = __builtin_amdgcn_wmma_f32_16x16x32_f16(false, a.v, false, b0, (short)0, acc0, false, false);
    acc1 = __builtin_amdgcn_wmma_f32_16x16x32_f16(false, a.v, false, b1, (short)0, acc1, false, false);
    acc2 = __builtin_amdgcn_wmma_f32_16x16x32_f16(false, a.v, false, b2, (short)0, acc2, false, false);
    acc3 = __builtin_amdgcn_wmma_f32_16x16x32_f16(false, a.v, false, b3, (short)0, acc3, false, false);
  }

  const int mb = tm * 16 + ((lane >> 4) << 3);
  v8f accs[4] = {acc0, acc1, acc2, acc3};
#pragma unroll
  for (int t = 0; t < 4; ++t) {
    U8F c; c.v = accs[t];
    const int n = (tn0 + t) * 16 + nh;
    float pev = 0.f;
    if (posenc) {
      // column-dependent frequency; row term applied per r below
      pev = __expf(-(float)(n & ~1) * (9.210340371976184f / 256.f)); // ln(1e4)/D
    }
#pragma unroll
    for (int r = 0; r < 8; ++r) {
      int m = mb + r;
      float v = c.e[r] + bias[n];
      if (resid)  v += resid[(size_t)m * N + n];
      if (posenc) {
        int s = m % S_;
        v += (n & 1) ? __cosf((float)s * pev) : __sinf((float)s * pev);
      }
      if (relu) v = fmaxf(v, 0.f);
      if (out32) out32[(size_t)m * N + n] = v;
      if (out16) out16[(size_t)m * N + n] = (_Float16)v;
    }
  }
}

// ---------------------------------------------------------------------------
// Banded attention: one wave per (batch, head, 16-query tile).
// Window N<=112 -> at most 8 key tiles (128-wide strip).
// Scores via WMMA, f32 softmax in LDS, P.V via WMMA with transposed V.
// qkv layout: [b*S+s, 768], q|k at col 0|256, head at +h*128 (q pre-scaled).
// vT layout: [((b*H+h)*HD + dim)*S + key].
// ---------------------------------------------------------------------------
__global__ __launch_bounds__(32)
void attn_kernel(const _Float16* __restrict__ qkv, const _Float16* __restrict__ vT,
                 _Float16* __restrict__ ctx, const int* __restrict__ Np) {
  __shared__ alignas(32) float    sc[16][128];
  __shared__ alignas(32) _Float16 sp[16][128];

  const int qt = blockIdx.x, h = blockIdx.y, b = blockIdx.z;
  const int Nwin = *Np;
  const int i0 = qt * 16;
  int lo = i0 - Nwin;
  int t0 = (lo > 0) ? (lo >> 4) : 0;
  if (qt - t0 + 1 > 8) t0 = qt - 7;          // clamp strip to 128 columns
  const int nt = qt - t0 + 1;
  const int W  = nt * 16;

  const int lane = threadIdx.x;
  const int nh  = lane & 15;
  const int kbA = (lane < 16) ? 0 : 16;
  const int kbB = (lane < 16) ? 0 : 16;
  const size_t base = (size_t)b * S_ * 768;

  // Preload Q A-fragments: head_dim 128 = 4 chunks of K=32 (16B vector loads)
  U16H aq[4];
  const _Float16* qrow = qkv + base + (size_t)(i0 + nh) * 768 + h * HD_;
#pragma unroll
  for (int cc = 0; cc < 4; ++cc) {
    const _Float16* ap = qrow + cc * 32 + kbA;
    aq[cc].h[0] = *(const v8h*)ap;
    aq[cc].h[1] = *(const v8h*)(ap + 8);
  }

  // Score strip: S = Q @ K^T, banded mask (K-fragments are 32B contiguous)
  for (int kt = t0; kt <= qt; ++kt) {
    v8f acc = {};
    const _Float16* krow = qkv + base + (size_t)(kt * 16 + nh) * 768 + 256 + h * HD_ + kbB;
#pragma unroll
    for (int cc = 0; cc < 4; ++cc) {
      v16h bk = *(const v16h*)(krow + cc * 32);
      acc = __builtin_amdgcn_wmma_f32_16x16x32_f16(false, aq[cc].v, false, bk,
                                                   (short)0, acc, false, false);
    }
    U8F cacc; cacc.v = acc;
    const int mb = (lane >> 4) << 3;
#pragma unroll
    for (int r = 0; r < 8; ++r) {
      int m = mb + r;
      int gi = i0 + m, gj = kt * 16 + nh;
      bool ok = (gj <= gi) && (gi - gj <= Nwin);
      sc[m][(kt - t0) * 16 + nh] = ok ? cacc.e[r] : -1e30f;
    }
  }
  __syncthreads();

  // Softmax per query row (lanes 0..15); write P f16, zero-pad strip to 128
  if (lane < 16) {
    float mx = -1e30f;
    for (int c2 = 0; c2 < W; ++c2) mx = fmaxf(mx, sc[lane][c2]);
    float sum = 0.f;
    for (int c2 = 0; c2 < W; ++c2) { float e = __expf(sc[lane][c2] - mx); sum += e; sc[lane][c2] = e; }
    float inv = 1.f / sum;
    for (int c2 = 0; c2 < 128; ++c2)
      sp[lane][c2] = (c2 < W) ? (_Float16)(sc[lane][c2] * inv) : (_Float16)0.f;
  }
  __syncthreads();

  // ctx = P @ V : 8 dim-tiles; K chunks of 32 keys; V fragments 32B contiguous
  const int nch = (W + 31) >> 5;
  const size_t vbase = ((size_t)(b * H_ + h) * HD_) * S_;
  for (int dt = 0; dt < 8; ++dt) {
    v8f acc = {};
    const _Float16* vrow = vT + vbase + (size_t)(dt * 16 + nh) * S_ + t0 * 16 + kbB;
    for (int cc = 0; cc < nch; ++cc) {
      U16H ap;
      const _Float16* pp = &sp[nh][cc * 32 + kbA];
      ap.h[0] = *(const v8h*)pp;
      ap.h[1] = *(const v8h*)(pp + 8);
      v16h bv = *(const v16h*)(vrow + cc * 32);   // keys contiguous in vT
      acc = __builtin_amdgcn_wmma_f32_16x16x32_f16(false, ap.v, false, bv,
                                                   (short)0, acc, false, false);
    }
    U8F cacc; cacc.v = acc;
    const int mb = (lane >> 4) << 3;
#pragma unroll
    for (int r = 0; r < 8; ++r) {
      int m = mb + r;
      ctx[((size_t)b * S_ + i0 + m) * D_ + h * HD_ + dt * 16 + nh] = (_Float16)cacc.e[r];
    }
  }
}

// ---------------------------------------------------------------------------
// LayerNorm over D=256: one wave per row (8 rows / 256-thread block).
// ---------------------------------------------------------------------------
__global__ __launch_bounds__(256)
void ln_kernel(const float* __restrict__ in, const float* __restrict__ g,
               const float* __restrict__ bta, float* __restrict__ out32,
               _Float16* __restrict__ out16) {
  const int wid  = threadIdx.x >> 5;
  const int lane = threadIdx.x & 31;
  const size_t row = (size_t)blockIdx.x * 8 + wid;
  const float* p = in + row * D_;
  float v[8], s = 0.f, s2 = 0.f;
#pragma unroll
  for (int i = 0; i < 8; ++i) { v[i] = p[lane + 32 * i]; s += v[i]; s2 += v[i] * v[i]; }
#pragma unroll
  for (int o = 16; o > 0; o >>= 1) { s += __shfl_xor(s, o, 32); s2 += __shfl_xor(s2, o, 32); }
  float mean = s * (1.f / 256.f);
  float var  = s2 * (1.f / 256.f) - mean * mean;
  float inv  = rsqrtf(var + 1e-5f);
#pragma unroll
  for (int i = 0; i < 8; ++i) {
    int c = lane + 32 * i;
    float o = (v[i] - mean) * inv * g[c] + bta[c];
    if (out32) out32[row * D_ + c] = o;
    if (out16) out16[row * D_ + c] = (_Float16)o;
  }
}

// ---------------------------------------------------------------------------
extern "C" void kernel_launch(void* const* d_in, const int* in_sizes, int n_in,
                              void* d_out, int out_size, void* d_ws, size_t ws_size,
                              hipStream_t stream) {
  (void)in_sizes; (void)n_in; (void)out_size; (void)ws_size;
  const float* src  = (const float*)d_in[0];
  const float* encw = (const float*)d_in[1];
  const float* encb = (const float*)d_in[2];
  const float* ipw  = (const float*)d_in[3];
  const float* ipb  = (const float*)d_in[4];
  const float* opw  = (const float*)d_in[5];
  const float* opb  = (const float*)d_in[6];
  const float* ln1g = (const float*)d_in[7];
  const float* ln1b = (const float*)d_in[8];
  const float* f1w  = (const float*)d_in[9];
  const float* f1b  = (const float*)d_in[10];
  const float* f2w  = (const float*)d_in[11];
  const float* f2b  = (const float*)d_in[12];
  const float* ln2g = (const float*)d_in[13];
  const float* ln2b = (const float*)d_in[14];
  const float* dcw  = (const float*)d_in[15];
  const float* dcb  = (const float*)d_in[16];
  const int*   Np   = (const int*)d_in[17];

  char* ws = (char*)d_ws;
  size_t off = 0;
  auto alloc = [&](size_t bytes) -> void* {
    void* p = ws + off;
    off += (bytes + 255) & ~(size_t)255;
    return p;
  };
  _Float16* enc_wp = (_Float16*)alloc((size_t)IN_ * D_ * 2);
  _Float16* ipw_p  = (_Float16*)alloc((size_t)D_ * 3 * D_ * 2);
  _Float16* opw_p  = (_Float16*)alloc((size_t)D_ * D_ * 2);
  _Float16* f1w_p  = (_Float16*)alloc((size_t)D_ * FF_ * 2);
  _Float16* f2w_p  = (_Float16*)alloc((size_t)FF_ * D_ * 2);
  _Float16* dcw_p  = (_Float16*)alloc((size_t)D_ * OUT_ * 2);
  float*    ipb_s  = (float*)alloc((size_t)3 * D_ * 4);
  _Float16* src_h  = (_Float16*)alloc((size_t)M_ * IN_ * 2);
  float*    x32    = (float*)alloc((size_t)M_ * D_ * 4);
  _Float16* x16    = (_Float16*)alloc((size_t)M_ * D_ * 2);
  _Float16* qkv16  = (_Float16*)alloc((size_t)M_ * 3 * D_ * 2);
  _Float16* vT16   = (_Float16*)alloc((size_t)M_ * D_ * 2);
  _Float16* ctx16  = (_Float16*)alloc((size_t)M_ * D_ * 2);
  float*    t32    = (float*)alloc((size_t)M_ * D_ * 4);
  float*    y132   = (float*)alloc((size_t)M_ * D_ * 4);
  _Float16* y116   = (_Float16*)alloc((size_t)M_ * D_ * 2);
  _Float16* u16    = (_Float16*)alloc((size_t)M_ * D_ * 2);
  _Float16* y216   = (_Float16*)alloc((size_t)M_ * D_ * 2);

  const float qscale = 0.08838834764831845f;  // 1/sqrt(hd=128)
  auto cdiv = [](int a, int b) { return (a + b - 1) / b; };

  // Weight prep -> fragment-packed f16 (q-scale folded into in_proj cols 0..255)
  wpack_kernel<<<cdiv(D_ * IN_, 256), 256, 0, stream>>>(encw, enc_wp, D_, IN_, 0, 1.f);
  wpack_kernel<<<cdiv(3 * D_ * D_, 256), 256, 0, stream>>>(ipw, ipw_p, 3 * D_, D_, D_, qscale);
  wpack_kernel<<<cdiv(D_ * D_, 256), 256, 0, stream>>>(opw, opw_p, D_, D_, 0, 1.f);
  wpack_kernel<<<cdiv(FF_ * D_, 256), 256, 0, stream>>>(f1w, f1w_p, FF_, D_, 0, 1.f);
  wpack_kernel<<<cdiv(D_ * FF_, 256), 256, 0, stream>>>(f2w, f2w_p, D_, FF_, 0, 1.f);
  wpack_kernel<<<cdiv(OUT_ * D_, 256), 256, 0, stream>>>(dcw, dcw_p, OUT_, D_, 0, 1.f);
  bscale_kernel<<<cdiv(3 * D_, 256), 256, 0, stream>>>(ipb, ipb_s, 3 * D_, D_, qscale);
  f32_to_f16_kernel<<<cdiv(M_ * IN_, 256), 256, 0, stream>>>(src, src_h, M_ * IN_);

  const int MT = M_ / 16;  // 2000 row tiles; grid.x = N/64 (4 n-tiles per wave)
  // Encode: x = src @ enc_w^T + b + pos_enc  -> f32 (residual) + f16
  wmma_gemm_kernel<<<dim3(D_ / 64, MT), 32, 0, stream>>>(src_h, enc_wp, encb, nullptr,
                                                         x32, x16, IN_, D_, 0, 1);
  // QKV (q pre-scaled): [M x 768] f16
  wmma_gemm_kernel<<<dim3(3 * D_ / 64, MT), 32, 0, stream>>>(x16, ipw_p, ipb_s, nullptr,
                                                             nullptr, qkv16, D_, 3 * D_, 0, 0);
  // V transpose for attention B-fragments
  vtrans_kernel<<<cdiv(M_ * D_, 256), 256, 0, stream>>>(qkv16, vT16);
  // Banded attention -> ctx f16
  attn_kernel<<<dim3(S_ / 16, H_, B_), 32, 0, stream>>>(qkv16, vT16, ctx16, Np);
  // out_proj + residual x -> t32
  wmma_gemm_kernel<<<dim3(D_ / 64, MT), 32, 0, stream>>>(ctx16, opw_p, opb, x32,
                                                         t32, nullptr, D_, D_, 0, 0);
  // LN1 -> y1 (f32 + f16)
  ln_kernel<<<M_ / 8, 256, 0, stream>>>(t32, ln1g, ln1b, y132, y116);
  // FF1 + relu -> u16
  wmma_gemm_kernel<<<dim3(FF_ / 64, MT), 32, 0, stream>>>(y116, f1w_p, f1b, nullptr,
                                                          nullptr, u16, D_, FF_, 1, 0);
  // FF2 + residual y1 -> t32
  wmma_gemm_kernel<<<dim3(D_ / 64, MT), 32, 0, stream>>>(u16, f2w_p, f2b, y132,
                                                         t32, nullptr, FF_, D_, 0, 0);
  // LN2 -> y2 f16
  ln_kernel<<<M_ / 8, 256, 0, stream>>>(t32, ln2g, ln2b, nullptr, y216);
  // Decode -> d_out f32 [M x 64]
  wmma_gemm_kernel<<<dim3(OUT_ / 64, MT), 32, 0, stream>>>(y216, dcw_p, dcb, nullptr,
                                                           (float*)d_out, nullptr, D_, OUT_, 0, 0);
}
